// BBPSeq2SeqAttn_26310969655896
// MI455X (gfx1250) — compile-verified, matrix operand order checked by reading, not run on hardware
//
#include <hip/hip_runtime.h>
#include <hip/hip_bf16.h>

// ---------------------------------------------------------------------------
// Problem constants (match reference)
// ---------------------------------------------------------------------------
#define D_   512
#define B_   64
#define T_   48
#define V_   32000
#define KT_  16          // K tiles: 512 / 32
#define MT_  4           // M tiles: 64 / 16
#define A_PANEL_BYTES (KT_ * 32 * 32)   // one mtile's packed A panel: 16 KB

typedef __attribute__((ext_vector_type(16))) __bf16 v16bf;
typedef __attribute__((ext_vector_type(8)))  float  v8f;
typedef __attribute__((ext_vector_type(4)))  float  v4f;
typedef int v4i __attribute__((ext_vector_type(4)));

// gfx1250 async global->LDS path (guarded; falls back to plain copy)
#if defined(__has_builtin)
#  if __has_builtin(__builtin_amdgcn_global_load_async_to_lds_b128)
#    define HAVE_ASYNC_LDS 1
#  endif
#endif
#ifndef HAVE_ASYNC_LDS
#  define HAVE_ASYNC_LDS 0
#endif

#if HAVE_ASYNC_LDS
// Signature (from hipcc diagnostic): (int4 AS1*, int4 AS3*, Imm int, Imm int)
typedef __attribute__((address_space(1))) v4i gv4i;
typedef __attribute__((address_space(3))) v4i lv4i;
#endif

__device__ __forceinline__ void wait_async_then_barrier() {
#if HAVE_ASYNC_LDS
#  if __has_builtin(__builtin_amdgcn_s_wait_asynccnt)
  __builtin_amdgcn_s_wait_asynccnt(0);
#  else
  asm volatile("s_wait_asynccnt 0x0" ::: "memory");
#  endif
#endif
  __syncthreads();
}

// ---------------------------------------------------------------------------
// Packed bf16 operand layouts (per CDNA5 ISA 7.12.2, 16-bit A 16x32 / B 32x16):
//   lane (0..31): row/col = lane&15, K-half select = lane>>4
//   16 bf16 slots per lane; slot(kk): ksel=(kk>>3)&1, p=((kk&7)>>1)+((kk>>4)<<2),
//   slot = 2p + (kk&1), lane = r + 16*ksel.
// Buffers are arrays of v16bf indexed ((tile*KT + kt)*32 + lane).
// ---------------------------------------------------------------------------
__device__ __forceinline__ void pack16_store(__hip_bfloat16* dst, int tile_rc,
                                             int r, int k, float val) {
  int kt   = k >> 5;
  int kk   = k & 31;
  int ksel = (kk >> 3) & 1;
  int p    = ((kk & 7) >> 1) + ((kk >> 4) << 2);
  int lane = r + (ksel << 4);
  int slot = (p << 1) + (kk & 1);
  size_t idx = ((((size_t)tile_rc * KT_ + kt) * 32 + lane) << 4) + slot;
  dst[idx] = __float2bfloat16(val);
}

// ---------------------------------------------------------------------------
// Cooperative LDS staging of one mtile's packed A panel (16 KB), 128 threads.
// Uses gfx1250 global_load_async_to_lds_b128 when available.
// ---------------------------------------------------------------------------
__device__ __forceinline__ void stage_A_panel(char* lds, const __hip_bfloat16* Ap,
                                              int mtile) {
  const int tid = threadIdx.y * 32 + threadIdx.x;         // 0..127
  const char* Ag = (const char*)Ap + (size_t)mtile * A_PANEL_BYTES;
#if HAVE_ASYNC_LDS
  #pragma unroll
  for (int i = 0; i < A_PANEL_BYTES / (128 * 16); ++i) {  // 8 x b128 per thread
    int ofs = (tid + i * 128) * 16;
    __builtin_amdgcn_global_load_async_to_lds_b128(
        (gv4i*)(Ag + ofs), (lv4i*)(lds + ofs), 0, 0);
  }
#else
  #pragma unroll
  for (int i = 0; i < A_PANEL_BYTES / (128 * 16); ++i) {
    int ofs = tid + i * 128;
    ((v4f*)lds)[ofs] = ((const v4f*)Ag)[ofs];
  }
#endif
  wait_async_then_barrier();
}

// ---------------------------------------------------------------------------
// One-wave 16x64 GEMM tile over K=512: A panel from LDS (shared by 4 waves),
// 4 N-tiles per wave -> each A fetch feeds 4 v_wmma_f32_16x16x32_bf16.
// ---------------------------------------------------------------------------
__device__ __forceinline__ void gemm_wave4(const char* lds_a,
                                           const __hip_bfloat16* Bp,
                                           const float* Cinit, float* C,
                                           int mtile, int ntile0, int ldc) {
  const int lane = threadIdx.x;          // 0..31
  const int ksel = lane >> 4;
  const int nn   = lane & 15;
  const v16bf* Al = reinterpret_cast<const v16bf*>(lds_a);
  const v16bf* B  = reinterpret_cast<const v16bf*>(Bp);

  v8f acc[4] = {};
  if (Cinit) {
    #pragma unroll
    for (int j = 0; j < 4; ++j)
      #pragma unroll
      for (int v = 0; v < 8; ++v)
        acc[j][v] = Cinit[(size_t)(mtile * 16 + v + 8 * ksel) * ldc +
                          (ntile0 + j) * 16 + nn];
  }

  #pragma unroll
  for (int kt = 0; kt < KT_; ++kt) {
    v16bf a = Al[kt * 32 + lane];                         // ds_load_b128 x2
    v16bf b0 = B[((size_t)(ntile0 + 0) * KT_ + kt) * 32 + lane];
    v16bf b1 = B[((size_t)(ntile0 + 1) * KT_ + kt) * 32 + lane];
    v16bf b2 = B[((size_t)(ntile0 + 2) * KT_ + kt) * 32 + lane];
    v16bf b3 = B[((size_t)(ntile0 + 3) * KT_ + kt) * 32 + lane];
    if (kt + 1 < KT_) {                                   // global_prefetch_b8
      __builtin_prefetch(&B[((size_t)(ntile0 + 0) * KT_ + kt + 1) * 32 + lane], 0, 3);
      __builtin_prefetch(&B[((size_t)(ntile0 + 1) * KT_ + kt + 1) * 32 + lane], 0, 3);
      __builtin_prefetch(&B[((size_t)(ntile0 + 2) * KT_ + kt + 1) * 32 + lane], 0, 3);
      __builtin_prefetch(&B[((size_t)(ntile0 + 3) * KT_ + kt + 1) * 32 + lane], 0, 3);
    }
    acc[0] = __builtin_amdgcn_wmma_f32_16x16x32_bf16(false, a, false, b0, (short)0, acc[0], false, false);
    acc[1] = __builtin_amdgcn_wmma_f32_16x16x32_bf16(false, a, false, b1, (short)0, acc[1], false, false);
    acc[2] = __builtin_amdgcn_wmma_f32_16x16x32_bf16(false, a, false, b2, (short)0, acc[2], false, false);
    acc[3] = __builtin_amdgcn_wmma_f32_16x16x32_bf16(false, a, false, b3, (short)0, acc[3], false, false);
  }

  #pragma unroll
  for (int j = 0; j < 4; ++j)
    #pragma unroll
    for (int v = 0; v < 8; ++v)
      C[(size_t)(mtile * 16 + v + 8 * ksel) * ldc + (ntile0 + j) * 16 + nn] = acc[j][v];
}

// ---------------------------------------------------------------------------
// Kernel: convert fp32 weight matrix to packed-bf16 B-operand layout.
// B[k][n] = tr ? W[n*K+k] (x @ W^T) : W[k*N+n] (h @ U)
// ---------------------------------------------------------------------------
__global__ void packB_k(const float* __restrict__ W, __hip_bfloat16* dst,
                        int N, int K, int tr) {
  size_t tid = (size_t)blockIdx.x * blockDim.x + threadIdx.x;
  if (tid >= (size_t)N * K) return;
  int n = (int)(tid / K);
  int k = (int)(tid - (size_t)n * K);
  float v = tr ? W[(size_t)n * K + k] : W[(size_t)k * N + n];
  pack16_store(dst, n >> 4, n & 15, k, v);
}

// Kernel: pack fp32 (64 x 512, row-major) activation into packed-A layout
__global__ void packA_k(const float* __restrict__ src, __hip_bfloat16* dst) {
  int tid = blockIdx.x * blockDim.x + threadIdx.x;   // 0..32767
  int m = tid >> 9, k = tid & 511;
  pack16_store(dst, m >> 4, m & 15, k, src[tid]);
}

// Kernel: embedding gather -> packed-A bf16 operand
__global__ void embed_k(const int* __restrict__ tok, const float* __restrict__ E,
                        __hip_bfloat16* xb, int t) {
  int tid = blockIdx.x * blockDim.x + threadIdx.x;   // 0..32767
  int m = tid >> 9, k = tid & 511;
  int token = tok[t * B_ + m];
  float v = E[(size_t)token * D_ + k];
  pack16_store(xb, m >> 4, m & 15, k, v);
}

// ---------------------------------------------------------------------------
// Kernel: the 4 GRU-cell GEMMs (h@U, x@Wz^T, x@Wr^T, x@Wn^T), fused by z.
// grid (Ntiles/16=2, Mtiles=4, 4), block (32,4): wave = 16x64 tile.
// ---------------------------------------------------------------------------
__global__ void cell_gemm_k(const __hip_bfloat16* __restrict__ hb,
                            const __hip_bfloat16* __restrict__ xb,
                            const __hip_bfloat16* __restrict__ Ub,
                            const __hip_bfloat16* __restrict__ Wzb,
                            const __hip_bfloat16* __restrict__ Wrb,
                            const __hip_bfloat16* __restrict__ Wnb,
                            float* hW, float* xz, float* xr, float* xn) {
  __shared__ __align__(16) char lds_a[A_PANEL_BYTES];
  int z = blockIdx.z;
  const __hip_bfloat16* A = (z == 0) ? hb : xb;
  const __hip_bfloat16* B = (z == 0) ? Ub : (z == 1) ? Wzb : (z == 2) ? Wrb : Wnb;
  float* C = (z == 0) ? hW : (z == 1) ? xz : (z == 2) ? xr : xn;
  stage_A_panel(lds_a, A, blockIdx.y);
  int ntile0 = (blockIdx.x * 4 + threadIdx.y) * 4;
  gemm_wave4(lds_a, B, nullptr, C, blockIdx.y, ntile0, D_);
}

// Generic GEMM over K=512: used for ctx@Wctx^T (+h) and logits.
// grid (ceil(Ntiles/16), Mtiles), block (32,4)
__global__ void gemm_k(const __hip_bfloat16* __restrict__ A,
                       const __hip_bfloat16* __restrict__ B,
                       const float* __restrict__ Cinit, float* C,
                       int Ntiles, int ldc) {
  __shared__ __align__(16) char lds_a[A_PANEL_BYTES];
  stage_A_panel(lds_a, A, blockIdx.y);
  int ntile0 = (blockIdx.x * 4 + threadIdx.y) * 4;
  if (ntile0 >= Ntiles) return;
  gemm_wave4(lds_a, B, Cinit, C, blockIdx.y, ntile0, ldc);
}

// ---------------------------------------------------------------------------
// Kernel: GRU gate fusion. h_new = (1-z)*h + z*n, writes fp32 h (in place OK)
// and packed-bf16 h for the next GEMM; optionally mirrors h into H[t].
// ---------------------------------------------------------------------------
__global__ void gate_k(const float* __restrict__ hW, const float* __restrict__ xz,
                       const float* __restrict__ xr, const float* __restrict__ xn,
                       const float* __restrict__ bz, const float* __restrict__ br,
                       const float* __restrict__ bn,
                       float* h, __hip_bfloat16* hb_pack, float* Hout) {
  int tid = blockIdx.x * blockDim.x + threadIdx.x;   // 0..32767
  int m = tid >> 9, k = tid & 511;
  float hw = hW[tid];
  float z  = 1.0f / (1.0f + __expf(-(xz[tid] + bz[k] + hw)));
  float r  = 1.0f / (1.0f + __expf(-(xr[tid] + br[k] + hw)));
  float n  = tanhf(xn[tid] + bn[k] + r * hw);
  float hv = h[tid];
  float out = (1.0f - z) * hv + z * n;
  h[tid] = out;
  if (Hout) Hout[tid] = out;
  pack16_store(hb_pack, m >> 4, m & 15, k, out);
}

// ---------------------------------------------------------------------------
// Kernel: attention (scores, mask, softmax, context) for one batch row.
// grid (64), block (256)
// ---------------------------------------------------------------------------
__global__ void attn_k(const float* __restrict__ H, const float* __restrict__ h,
                       const int* __restrict__ src, float* ctx,
                       __hip_bfloat16* ctxb) {
  const int b = blockIdx.x, tid = threadIdx.x;
  __shared__ float hs[D_];
  __shared__ float sc[T_];
  __shared__ float red[256];

  hs[tid]       = h[b * D_ + tid];
  hs[tid + 256] = h[b * D_ + tid + 256];
  __syncthreads();

  const float scale = 0.044194173824159216f;   // 1/sqrt(512)
  for (int t = 0; t < T_; ++t) {
    const float* Ht = H + ((size_t)t * B_ + b) * D_;
    float p = Ht[tid] * hs[tid] + Ht[tid + 256] * hs[tid + 256];
    red[tid] = p;
    __syncthreads();
    for (int s = 128; s > 0; s >>= 1) {
      if (tid < s) red[tid] += red[tid + s];
      __syncthreads();
    }
    if (tid == 0) {
      float v = red[0] * scale;
      if (src[t * B_ + b] == 0) v = -1e9f;     // PAD mask
      sc[t] = v;
    }
    __syncthreads();
  }

  if (tid == 0) {
    float mx = -1e30f;
    for (int t = 0; t < T_; ++t) mx = fmaxf(mx, sc[t]);
    float s = 0.0f;
    for (int t = 0; t < T_; ++t) { float e = __expf(sc[t] - mx); sc[t] = e; s += e; }
    float inv = 1.0f / s;
    for (int t = 0; t < T_; ++t) sc[t] *= inv;
  }
  __syncthreads();

  for (int d = tid; d < D_; d += 256) {
    float acc = 0.0f;
    for (int t = 0; t < T_; ++t)
      acc += sc[t] * H[((size_t)t * B_ + b) * D_ + d];
    ctx[b * D_ + d] = acc;
    pack16_store(ctxb, b >> 4, b & 15, d, acc);
  }
}

// ---------------------------------------------------------------------------
// Host orchestration
// ---------------------------------------------------------------------------
extern "C" void kernel_launch(void* const* d_in, const int* in_sizes, int n_in,
                              void* d_out, int out_size, void* d_ws, size_t ws_size,
                              hipStream_t stream) {
  const int*   src   = (const int*)  d_in[0];
  const int*   tgt   = (const int*)  d_in[1];
  const float* U     = (const float*)d_in[2];
  const float* Wz    = (const float*)d_in[3];
  const float* bz    = (const float*)d_in[4];
  const float* Wr    = (const float*)d_in[5];
  const float* br    = (const float*)d_in[6];
  const float* Wn    = (const float*)d_in[7];
  const float* bn    = (const float*)d_in[8];
  const float* E_en  = (const float*)d_in[9];
  const float* E_de  = (const float*)d_in[10];
  const float* Wout  = (const float*)d_in[11];
  const float* Wctx  = (const float*)d_in[12];
  float* logits = (float*)d_out;

  // ---- workspace carve-up -------------------------------------------------
  char* ws = (char*)d_ws;
  size_t off = 0;
  auto carve = [&](size_t bytes) { size_t r = off; off += (bytes + 255) & ~(size_t)255; return r; };
  const size_t sqB = (size_t)D_ * D_ * 2;           // one 512x512 bf16 matrix
  size_t oUb    = carve(sqB);
  size_t oWzb   = carve(sqB);
  size_t oWrb   = carve(sqB);
  size_t oWnb   = carve(sqB);
  size_t oWctxb = carve(sqB);
  size_t oWoutb = carve((size_t)V_ * D_ * 2);
  size_t oh1    = carve((size_t)B_ * D_ * 4);
  size_t oh2    = carve((size_t)B_ * D_ * 4);
  size_t oh1b   = carve((size_t)B_ * D_ * 2);
  size_t oh2b   = carve((size_t)B_ * D_ * 2);
  size_t ohW    = carve((size_t)B_ * D_ * 4);
  size_t oxz    = carve((size_t)B_ * D_ * 4);
  size_t oxr    = carve((size_t)B_ * D_ * 4);
  size_t oxn    = carve((size_t)B_ * D_ * 4);
  size_t oxb    = carve((size_t)B_ * D_ * 2);
  size_t oH     = carve((size_t)T_ * B_ * D_ * 4);
  size_t octx   = carve((size_t)B_ * D_ * 4);
  size_t octxb  = carve((size_t)B_ * D_ * 2);
  size_t oout   = carve((size_t)B_ * D_ * 4);
  size_t ooutb  = carve((size_t)B_ * D_ * 2);

  __hip_bfloat16* Ub    = (__hip_bfloat16*)(ws + oUb);
  __hip_bfloat16* Wzb   = (__hip_bfloat16*)(ws + oWzb);
  __hip_bfloat16* Wrb   = (__hip_bfloat16*)(ws + oWrb);
  __hip_bfloat16* Wnb   = (__hip_bfloat16*)(ws + oWnb);
  __hip_bfloat16* Wctxb = (__hip_bfloat16*)(ws + oWctxb);
  __hip_bfloat16* Woutb = (__hip_bfloat16*)(ws + oWoutb);
  float* h1   = (float*)(ws + oh1);
  float* h2   = (float*)(ws + oh2);
  __hip_bfloat16* h1b = (__hip_bfloat16*)(ws + oh1b);
  __hip_bfloat16* h2b = (__hip_bfloat16*)(ws + oh2b);
  float* hW   = (float*)(ws + ohW);
  float* xz   = (float*)(ws + oxz);
  float* xr   = (float*)(ws + oxr);
  float* xn   = (float*)(ws + oxn);
  __hip_bfloat16* xb = (__hip_bfloat16*)(ws + oxb);
  float* H    = (float*)(ws + oH);
  float* ctx  = (float*)(ws + octx);
  __hip_bfloat16* ctxb = (__hip_bfloat16*)(ws + octxb);
  float* outf = (float*)(ws + oout);
  __hip_bfloat16* outb = (__hip_bfloat16*)(ws + ooutb);

  // ---- one-time weight conversion to packed bf16 --------------------------
  {
    int thr = 256;
    int nsq = (D_ * D_ + thr - 1) / thr;
    packB_k<<<nsq, thr, 0, stream>>>(U,    Ub,    D_, D_, 0);  // h @ U
    packB_k<<<nsq, thr, 0, stream>>>(Wz,   Wzb,   D_, D_, 1);  // x @ Wz^T
    packB_k<<<nsq, thr, 0, stream>>>(Wr,   Wrb,   D_, D_, 1);
    packB_k<<<nsq, thr, 0, stream>>>(Wn,   Wnb,   D_, D_, 1);
    packB_k<<<nsq, thr, 0, stream>>>(Wctx, Wctxb, D_, D_, 1);  // ctx @ Wctx^T
    size_t nv = (size_t)V_ * D_;
    packB_k<<<(unsigned)((nv + thr - 1) / thr), thr, 0, stream>>>(Wout, Woutb, V_, D_, 1);
  }

  // ---- zero initial hidden states (fp32 and packed bf16) ------------------
  (void)hipMemsetAsync(h1,  0, (size_t)B_ * D_ * 4, stream);
  (void)hipMemsetAsync(h2,  0, (size_t)B_ * D_ * 4, stream);
  (void)hipMemsetAsync(h1b, 0, (size_t)B_ * D_ * 2, stream);
  (void)hipMemsetAsync(h2b, 0, (size_t)B_ * D_ * 2, stream);

  const dim3 gBlk(32, 4, 1);
  const dim3 gCell(2, MT_, 4);       // 32 N-tiles (16/block) x 4 M-tiles x 4 mats
  const int  eThr = 256, eBlkN = (B_ * D_) / eThr;   // 128 blocks

  // ---- encoder ------------------------------------------------------------
  for (int t = 0; t < T_; ++t) {
    embed_k<<<eBlkN, eThr, 0, stream>>>(src, E_en, xb, t);
    // layer 1: h1 = cell(x, h1)
    cell_gemm_k<<<gCell, gBlk, 0, stream>>>(h1b, xb, Ub, Wzb, Wrb, Wnb, hW, xz, xr, xn);
    gate_k<<<eBlkN, eThr, 0, stream>>>(hW, xz, xr, xn, bz, br, bn, h1, h1b, nullptr);
    // layer 2: h2 = cell(h1, h2); mirror into H[t]
    cell_gemm_k<<<gCell, gBlk, 0, stream>>>(h2b, h1b, Ub, Wzb, Wrb, Wnb, hW, xz, xr, xn);
    gate_k<<<eBlkN, eThr, 0, stream>>>(hW, xz, xr, xn, bz, br, bn, h2, h2b,
                                       H + (size_t)t * B_ * D_);
  }

  // ---- decoder (h continues from h2) --------------------------------------
  for (int t = 0; t < T_; ++t) {
    embed_k<<<eBlkN, eThr, 0, stream>>>(tgt, E_de, xb, t);
    cell_gemm_k<<<gCell, gBlk, 0, stream>>>(h2b, xb, Ub, Wzb, Wrb, Wnb, hW, xz, xr, xn);
    gate_k<<<eBlkN, eThr, 0, stream>>>(hW, xz, xr, xn, bz, br, bn, h2, h2b, nullptr);

    attn_k<<<B_, 256, 0, stream>>>(H, h2, src, ctx, ctxb);

    // out = h + ctx @ Wctx^T   (Cinit = h2)
    gemm_k<<<dim3(2, MT_), gBlk, 0, stream>>>(ctxb, Wctxb, h2, outf, D_ / 16, D_);
    packA_k<<<eBlkN, eThr, 0, stream>>>(outf, outb);

    // logits[t] = out @ Wout^T  (64 x 32000), 125 blocks x 4 waves x 4 ntiles
    gemm_k<<<dim3(125, MT_), gBlk, 0, stream>>>(outb, Woutb, nullptr,
                                                logits + (size_t)t * B_ * V_,
                                                V_ / 16, V_);
  }
}
